// MultiHeadSelfAttention_17179869293
// MI455X (gfx1250) — matrix operand order repaired and checked by reading, not compiled, not run on hardware
//
#include <hip/hip_runtime.h>

typedef __attribute__((ext_vector_type(16))) __bf16 v16bf;
typedef __attribute__((ext_vector_type(8)))  __bf16 v8bf;
typedef __attribute__((ext_vector_type(8)))  float  v8f;

#define BN    4096   // B*N
#define NSEQ  2048
#define CDIM  1024
#define NH    16
#define HD    64

union ABFrag { v16bf v; v8bf h[2]; };

__device__ __forceinline__ v8f vzero() {
  v8f z;
#pragma unroll
  for (int i = 0; i < 8; ++i) z[i] = 0.0f;
  return z;
}

__device__ __forceinline__ v8f wmma_bf16(v16bf a, v16bf b, v8f c) {
  // 8 args: (neg_a, A, neg_b, B, c_mod, C, reuse_a, reuse_b)
  return __builtin_amdgcn_wmma_f32_16x16x32_bf16(false, a, false, b, (short)0, c,
                                                 false, false);
}

// A fragment (16x32, bf16): row-major source, rows = M dim, cols = K dim.
// Lane l holds row (l&15); halves h=0..15 -> K = k0+hi*16+h, contiguous.
__device__ __forceinline__ v16bf load_a_frag(const __bf16* base, int lda,
                                             int row0, int k0, int lane) {
  int lr = lane & 15, hi = lane >> 4;
  const __bf16* p = base + (size_t)(row0 + lr) * lda + (k0 + hi * 16);
  ABFrag u;
  u.h[0] = *(const v8bf*)(p);
  u.h[1] = *(const v8bf*)(p + 8);
  return u.v;
}

// B fragment (32x16, bf16) where Bmat[k][n] = src[n0+n][k0+k] (K-contiguous
// source, i.e. src holds B transposed). Lane l: N = l&15, halves h=0..15 ->
// K = hi*16 + h, contiguous in memory.
__device__ __forceinline__ v16bf load_bt_frag(const __bf16* base, int ldb,
                                              int n0, int k0, int lane) {
  int ln = lane & 15, hi = lane >> 4;
  const __bf16* p = base + (size_t)(n0 + ln) * ldb + (k0 + hi * 16);
  ABFrag u;
  u.h[0] = *(const v8bf*)(p);
  u.h[1] = *(const v8bf*)(p + 8);
  return u.v;
}

// Software-pipelined 32(M) x 64(N) x CDIM(K) GEMM mainloop:
//   acc[mt][c] += A[row0+mt*16 .., k] * Bsrc[col0+c*16 ..][k]
// Next k-step's fragments are prefetched (wrap-around keeps one basic block)
// before the current WMMAs issue, so loads overlap the matrix pipe.
__device__ __forceinline__ void gemm_tile_32x64(
    const __bf16* __restrict__ A, int lda, const __bf16* __restrict__ Bm,
    int ldb, int row0, int col0, int lane, v8f acc[2][4]) {
#pragma unroll
  for (int mt = 0; mt < 2; ++mt)
#pragma unroll
    for (int c = 0; c < 4; ++c) acc[mt][c] = vzero();

  v16bf a0 = load_a_frag(A, lda, row0, 0, lane);
  v16bf a1 = load_a_frag(A, lda, row0 + 16, 0, lane);
  v16bf bf[4];
#pragma unroll
  for (int c = 0; c < 4; ++c) bf[c] = load_bt_frag(Bm, ldb, col0 + c * 16, 0, lane);

  for (int k0 = 0; k0 < CDIM; k0 += 32) {
    int kn = (k0 + 32) & (CDIM - 1);   // wrap: last prefetch is harmless
    v16bf a0n = load_a_frag(A, lda, row0, kn, lane);
    v16bf a1n = load_a_frag(A, lda, row0 + 16, kn, lane);
    v16bf bfn[4];
#pragma unroll
    for (int c = 0; c < 4; ++c)
      bfn[c] = load_bt_frag(Bm, ldb, col0 + c * 16, kn, lane);
#pragma unroll
    for (int c = 0; c < 4; ++c) acc[0][c] = wmma_bf16(a0, bf[c], acc[0][c]);
#pragma unroll
    for (int c = 0; c < 4; ++c) acc[1][c] = wmma_bf16(a1, bf[c], acc[1][c]);
    a0 = a0n; a1 = a1n;
#pragma unroll
    for (int c = 0; c < 4; ++c) bf[c] = bfn[c];
  }
}

// ---------------------------------------------------------------- convert ---
__global__ void cvt_kernel(const float* __restrict__ in, __bf16* __restrict__ out,
                           int n4) {
  int i = blockIdx.x * blockDim.x + threadIdx.x;
  if (i < n4) {
    float4 f = ((const float4*)in)[i];
    out[i * 4 + 0] = (__bf16)f.x;
    out[i * 4 + 1] = (__bf16)f.y;
    out[i * 4 + 2] = (__bf16)f.z;
    out[i * 4 + 3] = (__bf16)f.w;
  }
}

// ---------------------------------------------------------------- QKV GEMM ---
// y[BN,3C] = xb @ wqkv^T + bias; epilogue scatters into Q (scaled), K, V^T.
__global__ void __launch_bounds__(256) qkv_kernel(
    const __bf16* __restrict__ xb, const __bf16* __restrict__ wq,
    const float* __restrict__ bias, __bf16* __restrict__ Q,
    __bf16* __restrict__ K, __bf16* __restrict__ Vt) {
  int lane = threadIdx.x & 31;
  int wid  = threadIdx.x >> 5;
  int t    = blockIdx.x * 8 + wid;        // 0..6143
  int mt = t / 48, nt = t % 48;
  int row0 = mt * 32, col0 = nt * 64;
  int ln = lane & 15, hi = lane >> 4;

  v8f acc[2][4];
  gemm_tile_32x64(xb, CDIM, wq, CDIM, row0, col0, lane, acc);

#pragma unroll
  for (int c = 0; c < 4; ++c) {
    int m   = col0 + c * 16 + ln;          // output feature 0..3071
    int sel = m >> 10;                     // 0=Q 1=K 2=V (wave-uniform)
    int f   = m & 1023;
    int hh  = f >> 6, hd = f & 63;
    float bv = bias[m];
#pragma unroll
    for (int mt2 = 0; mt2 < 2; ++mt2) {
#pragma unroll
      for (int r = 0; r < 8; ++r) {
        int ng = row0 + mt2 * 16 + r + hi * 8;  // 0..4095 (= b*N + n)
        int bb = ng >> 11, nn = ng & 2047;
        float val = acc[mt2][c][r] + bv;
        if (sel == 0)
          Q[((size_t)(bb * NH + hh) * NSEQ + nn) * HD + hd] =
              (__bf16)(val * 0.125f);
        else if (sel == 1)
          K[((size_t)(bb * NH + hh) * NSEQ + nn) * HD + hd] = (__bf16)val;
        else
          Vt[((size_t)(bb * NH + hh) * HD + hd) * NSEQ + nn] = (__bf16)val;
      }
    }
  }
}

// ---------------------------------------------------------- flash attention ---
// One wave per 16-row query tile; stream 32 keys per iteration. K fragments
// for the next tile and V fragments for this tile are issued before the
// VALU-heavy online softmax so global loads overlap exp/shuffle work.
__global__ void __launch_bounds__(256) attn_kernel(
    const __bf16* __restrict__ Q, const __bf16* __restrict__ K,
    const __bf16* __restrict__ Vt, __bf16* __restrict__ attn) {
  __shared__ __attribute__((aligned(32))) __bf16 pbuf[8][16 * 32];
  int lane = threadIdx.x & 31;
  int wid  = threadIdx.x >> 5;
  int gw   = blockIdx.x * 8 + wid;         // 0..4095
  int bh = gw >> 7, qt = gw & 127;
  int q0 = qt * 16;
  const __bf16* Qp = Q  + (size_t)bh * NSEQ * HD;
  const __bf16* Kp = K  + (size_t)bh * NSEQ * HD;
  const __bf16* Vp = Vt + (size_t)bh * HD * NSEQ;
  int ln = lane & 15, hi = lane >> 4;
  __bf16* pb = &pbuf[wid][0];

  v16bf aq0 = load_a_frag(Qp, HD, q0, 0, lane);
  v16bf aq1 = load_a_frag(Qp, HD, q0, 32, lane);

  v8f o[4];
#pragma unroll
  for (int c = 0; c < 4; ++c) o[c] = vzero();
  float mrow[8], lrow[8];
#pragma unroll
  for (int r = 0; r < 8; ++r) { mrow[r] = -1e30f; lrow[r] = 0.0f; }

  // K fragments for key pair 0: [tile 0 ks=0, tile 0 ks=32, tile 1 ks=0, ks=32]
  v16bf kb[4];
  kb[0] = load_bt_frag(Kp, HD, 0,  0,  lane);
  kb[1] = load_bt_frag(Kp, HD, 0,  32, lane);
  kb[2] = load_bt_frag(Kp, HD, 16, 0,  lane);
  kb[3] = load_bt_frag(Kp, HD, 16, 32, lane);

  for (int kt = 0; kt < NSEQ / 32; ++kt) {
    int key0 = kt * 32;
    // S tiles for keys [key0, key0+16) and [key0+16, key0+32)
    v8f s0 = vzero(), s1 = vzero();
    s0 = wmma_bf16(aq0, kb[0], s0);
    s0 = wmma_bf16(aq1, kb[1], s0);
    s1 = wmma_bf16(aq0, kb[2], s1);
    s1 = wmma_bf16(aq1, kb[3], s1);

    // issue V loads for this tile + K loads for next tile (wrap: harmless)
    v16bf vb[4];
#pragma unroll
    for (int c = 0; c < 4; ++c)
      vb[c] = load_bt_frag(Vp, NSEQ, c * 16, key0, lane);
    int keyn = (key0 + 32) & (NSEQ - 1);
    v16bf kbn[4];
    kbn[0] = load_bt_frag(Kp, HD, keyn,      0,  lane);
    kbn[1] = load_bt_frag(Kp, HD, keyn,      32, lane);
    kbn[2] = load_bt_frag(Kp, HD, keyn + 16, 0,  lane);
    kbn[3] = load_bt_frag(Kp, HD, keyn + 16, 32, lane);

    // online softmax per row (rows of tile live in 16-lane half-groups)
#pragma unroll
    for (int r = 0; r < 8; ++r) {
      float mx = fmaxf(s0[r], s1[r]);
      mx = fmaxf(mx, __shfl_xor(mx, 1, 32));
      mx = fmaxf(mx, __shfl_xor(mx, 2, 32));
      mx = fmaxf(mx, __shfl_xor(mx, 4, 32));
      mx = fmaxf(mx, __shfl_xor(mx, 8, 32));
      float mnew  = fmaxf(mrow[r], mx);
      float scale = __expf(mrow[r] - mnew);
      float p0 = __expf(s0[r] - mnew);
      float p1 = __expf(s1[r] - mnew);
      float ps = p0 + p1;
      ps += __shfl_xor(ps, 1, 32);
      ps += __shfl_xor(ps, 2, 32);
      ps += __shfl_xor(ps, 4, 32);
      ps += __shfl_xor(ps, 8, 32);
      lrow[r] = lrow[r] * scale + ps;
      mrow[r] = mnew;
      o[0][r] *= scale; o[1][r] *= scale; o[2][r] *= scale; o[3][r] *= scale;
      int M = r + hi * 8;
      pb[M * 32 + ln]      = (__bf16)p0;   // keys key0 .. +15
      pb[M * 32 + 16 + ln] = (__bf16)p1;   // keys key0+16 .. +31
    }
    // P (16x32) through LDS to re-swizzle into A-fragment layout
    v16bf pa = load_a_frag(pb, 32, 0, 0, lane);
#pragma unroll
    for (int c = 0; c < 4; ++c) o[c] = wmma_bf16(pa, vb[c], o[c]);
#pragma unroll
    for (int c = 0; c < 4; ++c) kb[c] = kbn[c];
  }

  int b_ = bh >> 4, h_ = bh & 15;
#pragma unroll
  for (int r = 0; r < 8; ++r) {
    float inv = 1.0f / lrow[r];
    int ng = q0 + r + hi * 8;
    size_t rowoff = ((size_t)b_ * NSEQ + ng) * CDIM + h_ * HD;
#pragma unroll
    for (int c = 0; c < 4; ++c)
      attn[rowoff + c * 16 + ln] = (__bf16)(o[c][r] * inv);
  }
}

// ------------------------------------------------------------- projection ---
__global__ void __launch_bounds__(256) proj_kernel(
    const __bf16* __restrict__ ab, const __bf16* __restrict__ wp,
    const float* __restrict__ bias, float* __restrict__ out) {
  int lane = threadIdx.x & 31;
  int wid  = threadIdx.x >> 5;
  int t    = blockIdx.x * 8 + wid;        // 0..2047
  int mt = t >> 4, nt = t & 15;
  int row0 = mt * 32, col0 = nt * 64;
  int ln = lane & 15, hi = lane >> 4;

  v8f acc[2][4];
  gemm_tile_32x64(ab, CDIM, wp, CDIM, row0, col0, lane, acc);

#pragma unroll
  for (int c = 0; c < 4; ++c) {
    int m = col0 + c * 16 + ln;
    float bv = bias[m];
#pragma unroll
    for (int mt2 = 0; mt2 < 2; ++mt2) {
#pragma unroll
      for (int r = 0; r < 8; ++r) {
        int ng = row0 + mt2 * 16 + r + hi * 8;
        out[(size_t)ng * CDIM + m] = acc[mt2][c][r] + bv;
      }
    }
  }
}

// ------------------------------------------------------------------ launch ---
extern "C" void kernel_launch(void* const* d_in, const int* in_sizes, int n_in,
                              void* d_out, int out_size, void* d_ws, size_t ws_size,
                              hipStream_t stream) {
  (void)in_sizes; (void)n_in; (void)out_size; (void)ws_size;
  const float* x      = (const float*)d_in[0];
  const float* qkv_w  = (const float*)d_in[1];
  const float* qkv_b  = (const float*)d_in[2];
  const float* proj_w = (const float*)d_in[3];
  const float* proj_b = (const float*)d_in[4];
  float* out = (float*)d_out;

  __bf16* ws    = (__bf16*)d_ws;
  __bf16* xb    = ws;                                   // [BN, C]
  __bf16* wq    = xb  + (size_t)BN * CDIM;              // [3C, C]
  __bf16* wp    = wq  + (size_t)3 * CDIM * CDIM;        // [C, C]
  __bf16* Qb    = wp  + (size_t)CDIM * CDIM;            // [B,H,N,HD] (pre-scaled)
  __bf16* Kb    = Qb  + (size_t)BN * CDIM;              // [B,H,N,HD]
  __bf16* Vtb   = Kb  + (size_t)BN * CDIM;              // [B,H,HD,N]
  __bf16* attnb = Vtb + (size_t)BN * CDIM;              // [B,N,C]

  cvt_kernel<<<(BN * CDIM / 4) / 256, 256, 0, stream>>>(x, xb, BN * CDIM / 4);
  cvt_kernel<<<(3 * CDIM * CDIM / 4) / 256, 256, 0, stream>>>(qkv_w, wq,
                                                              3 * CDIM * CDIM / 4);
  cvt_kernel<<<(CDIM * CDIM / 4) / 256, 256, 0, stream>>>(proj_w, wp,
                                                          CDIM * CDIM / 4);
  // (BN/32)*(3C/64) = 128*48 = 6144 waves / 8 per block
  qkv_kernel<<<768, 256, 0, stream>>>(xb, wq, qkv_b, Qb, Kb, Vtb);
  // B*H*(N/16) = 4096 waves / 8 per block
  attn_kernel<<<512, 256, 0, stream>>>(Qb, Kb, Vtb, attnb);
  // (BN/32)*(C/64) = 128*16 = 2048 waves / 8 per block
  proj_kernel<<<256, 256, 0, stream>>>(attnb, wp, proj_b, out);
}
